// SlidingWindowSinkAttention_4440996184665
// MI455X (gfx1250) — matrix operand order repaired and checked by reading, not compiled, not run on hardware
//
#include <hip/hip_runtime.h>
#include <hip/hip_bf16.h>

typedef __attribute__((ext_vector_type(16))) __bf16 bf16x16;
typedef __attribute__((ext_vector_type(8)))  __bf16 bf16x8;
typedef __attribute__((ext_vector_type(8)))  float  f32x8;
typedef __attribute__((ext_vector_type(4)))  float  f32x4;
typedef __attribute__((ext_vector_type(4)))  unsigned int u32x4;
typedef __attribute__((ext_vector_type(8)))  int    i32x8;
typedef __attribute__((ext_vector_type(4)))  int    i32x4;

#define N_HEADS   16
#define N_KV_HEADS 4
#define HEAD_DIM  128
#define D_MODEL   2048
#define SINK      4
#define WINDOW    256
#define BATCH     2
#define SEQ       2048
#define QKV_LD    3072          // 2048 Q | 512 K | 512 V per token row
#define MTOT      (BATCH*SEQ)   // 4096
#define LDS_STR   72            // 64 bf16 + 8 bf16 pad (TDM pads 16B per 128B row)

__device__ __forceinline__ f32x8 wmma_bf16(bf16x16 a, bf16x16 b, f32x8 c) {
  return __builtin_amdgcn_wmma_f32_16x16x32_bf16(false, a, false, b, (short)0, c, false, false);
}

// A-operand fragment: 16(M) x 32(K) bf16 from row-major [m][k], row stride ldm elements.
__device__ __forceinline__ bf16x16 load_fragA(const __bf16* base, int ldm, int lane) {
  const int r  = lane & 15;
  const int kb = (lane >> 4) << 3;      // 0 or 8
  const __bf16* p = base + r * ldm + kb;
  bf16x8 lo = *(const bf16x8*)(p);
  bf16x8 hi = *(const bf16x8*)(p + 16);
  return __builtin_shufflevector(lo, hi, 0,1,2,3,4,5,6,7,8,9,10,11,12,13,14,15);
}

// B-operand fragment: 32(K) x 16(N) bf16 sourced from B^T row-major [n][k], stride ldm.
__device__ __forceinline__ bf16x16 load_fragB(const __bf16* base, int ldm, int lane) {
  const int n  = lane & 15;
  const int kb = (lane >> 4) << 3;      // 0 or 8
  const __bf16* p = base + n * ldm + kb;
  bf16x8 lo = *(const bf16x8*)(p);
  bf16x8 hi = *(const bf16x8*)(p + 16);
  return __builtin_shufflevector(lo, hi, 0,1,2,3,4,5,6,7,8,9,10,11,12,13,14,15);
}

template<typename T> __device__ __forceinline__ T cvt_out(float v);
template<> __device__ __forceinline__ float  cvt_out<float>(float v)  { return v; }
template<> __device__ __forceinline__ __bf16 cvt_out<__bf16>(float v) { return (__bf16)v; }

// ---- Tensor Data Mover: DMA one 128-row x 64-col bf16 tile into LDS ----
// Pads 16B after every 128B row -> LDS row stride = 144B = 72 elements.
__device__ __forceinline__ void tdm_load_tile(const __bf16* gptr, unsigned ldsAddr,
                                              int rowsTotal, int K) {
  unsigned long long ga = (unsigned long long)(size_t)gptr;
  u32x4 g0;
  g0[0] = 1u;                                             // count=1 (valid), user mode
  g0[1] = ldsAddr;                                        // LDS byte address
  g0[2] = (unsigned)(ga & 0xFFFFFFFFull);                 // global_addr[31:0]
  g0[3] = (unsigned)((ga >> 32) & 0x1FFFFFFull) | (2u << 30); // addr[56:32] | type=2
  i32x8 g1;
  g1[0] = (1 << 16)       // data_size = 2 bytes
        | (1 << 20)       // pad_enable
        | (4 << 22)       // pad_interval: 32 DWORDs (128B) between pads
        | (3 << 25);      // pad_amount:   4 DWORDs (16B)
  g1[1] = (K & 0xFFFF) << 16;                             // tensor_dim0[15:0]
  g1[2] = ((K >> 16) & 0xFFFF) | ((rowsTotal & 0xFFFF) << 16);
  g1[3] = ((rowsTotal >> 16) & 0xFFFF) | (64 << 16);      // tile_dim0 = 64
  g1[4] = 128;                                            // tile_dim1 = 128
  g1[5] = K;                                              // tensor_dim0_stride[31:0]
  g1[6] = 0;
  g1[7] = 0;
  i32x4 z4 = {0, 0, 0, 0};
  i32x8 z8i = {0, 0, 0, 0, 0, 0, 0, 0};
  __builtin_amdgcn_tensor_load_to_lds(g0, g1, z4, z4, z8i, 0);
}

// f32 -> bf16 bulk convert (8 elements / thread)
__global__ __launch_bounds__(256)
void cvt_bf16(const float* __restrict__ src, __bf16* __restrict__ dst, int n8) {
  const int i = blockIdx.x * 256 + threadIdx.x;
  if (i >= n8) return;
  f32x4 a = *(const f32x4*)(src + (size_t)i * 8);
  f32x4 b = *(const f32x4*)(src + (size_t)i * 8 + 4);
  bf16x8 o;
  #pragma unroll
  for (int j = 0; j < 4; j++) { o[j] = (__bf16)a[j]; o[4 + j] = (__bf16)b[j]; }
  *(bf16x8*)(dst + (size_t)i * 8) = o;
}

// C[M,N](TOUT) = A[M,K](bf16) @ B[N,K](bf16)^T
// 128x128x64 tiles, TDM double-buffered LDS, 8 waves (2x4), wave = 64x32 of C.
template<typename TOUT>
__global__ __launch_bounds__(256)
void gemm_tdm(const __bf16* __restrict__ A, const __bf16* __restrict__ B,
              int M, int N, int K, TOUT* __restrict__ C) {
  __shared__ __bf16 As[2][128 * LDS_STR];
  __shared__ __bf16 Bs[2][128 * LDS_STR];
  const int tid = threadIdx.x, lane = tid & 31, wid = tid >> 5;
  const int wm = wid >> 2, wn = wid & 3;
  const int nTile = blockIdx.x * 128, mTile = blockIdx.y * 128;

  const f32x8 z8 = {0.f,0.f,0.f,0.f,0.f,0.f,0.f,0.f};
  f32x8 acc[4][2];
  #pragma unroll
  for (int i = 0; i < 4; i++) { acc[i][0] = z8; acc[i][1] = z8; }

  const __bf16* aBase = A + (size_t)mTile * K;
  const __bf16* bBase = B + (size_t)nTile * K;
  const unsigned ldsA[2] = { (unsigned)(size_t)&As[0][0], (unsigned)(size_t)&As[1][0] };
  const unsigned ldsB[2] = { (unsigned)(size_t)&Bs[0][0], (unsigned)(size_t)&Bs[1][0] };

  if (wid == 0) {
    tdm_load_tile(aBase, ldsA[0], M, K);
    tdm_load_tile(bBase, ldsB[0], N, K);
  }

  const int KSTEPS = K >> 6;
  for (int ks = 0; ks < KSTEPS; ks++) {
    const int cur = ks & 1, nxt = cur ^ 1;
    if (wid == 0) {
      if (ks + 1 < KSTEPS) {
        tdm_load_tile(aBase + (ks + 1) * 64, ldsA[nxt], M, K);
        tdm_load_tile(bBase + (ks + 1) * 64, ldsB[nxt], N, K);
        __builtin_amdgcn_s_wait_tensorcnt(2);   // current tile's 2 DMAs done (in-order)
      } else {
        __builtin_amdgcn_s_wait_tensorcnt(0);
      }
    }
    __syncthreads();                            // cur tile visible to all waves
    #pragma unroll
    for (int kc = 0; kc < 2; kc++) {
      bf16x16 b0 = load_fragB(&Bs[cur][(wn * 32)      * LDS_STR + kc * 32], LDS_STR, lane);
      bf16x16 b1 = load_fragB(&Bs[cur][(wn * 32 + 16) * LDS_STR + kc * 32], LDS_STR, lane);
      #pragma unroll
      for (int i = 0; i < 4; i++) {
        bf16x16 a = load_fragA(&As[cur][(wm * 64 + i * 16) * LDS_STR + kc * 32], LDS_STR, lane);
        acc[i][0] = wmma_bf16(a, b0, acc[i][0]);
        acc[i][1] = wmma_bf16(a, b1, acc[i][1]);
      }
    }
    __syncthreads();                            // all waves done reading cur before overwrite
  }

  const int hlf = lane >> 4, col = lane & 15;
  #pragma unroll
  for (int i = 0; i < 4; i++)
    #pragma unroll
    for (int j = 0; j < 2; j++) {
      const int n     = nTile + wn * 32 + j * 16 + col;
      const int mbase = mTile + wm * 64 + i * 16 + hlf * 8;
      #pragma unroll
      for (int r = 0; r < 8; r++)
        C[(size_t)(mbase + r) * N + n] = cvt_out<TOUT>(acc[i][j][r]);
    }
}

// vT[b][kvh][d][s] = V[b][s][kvh][d]
__global__ __launch_bounds__(256)
void transpose_v(const __bf16* __restrict__ qkv, __bf16* __restrict__ vT) {
  const int idx = blockIdx.x * 256 + threadIdx.x;
  const int s   = idx & (SEQ - 1);
  const int d   = (idx >> 11) & (HEAD_DIM - 1);
  const int kvh = (idx >> 18) & (N_KV_HEADS - 1);
  const int b   = idx >> 20;
  vT[idx] = qkv[(size_t)(b * SEQ + s) * QKV_LD + (D_MODEL + 512) + kvh * HEAD_DIM + d];
}

// One wave per (b, head, 16-query tile); flash attention over 32-key tiles + sink tile.
__global__ __launch_bounds__(256)
void attn_kernel(const __bf16* __restrict__ qkv, const __bf16* __restrict__ vT,
                 __bf16* __restrict__ attnO) {
  __shared__ __bf16 Plds[8][16 * 40];
  const int lane = threadIdx.x & 31;
  const int wid  = threadIdx.x >> 5;
  const int gw   = blockIdx.x * 8 + wid;     // 0..4095
  const int qt   = gw & 127;
  const int h    = (gw >> 7) & (N_HEADS - 1);
  const int b    = gw >> 11;
  const int kvh  = h >> 2;
  const int m0   = qt * 16;
  const int hlf  = lane >> 4, colb = lane & 15;
  __bf16* pl = &Plds[wid][0];

  const __bf16* qbase = qkv + (size_t)(b * SEQ + m0) * QKV_LD + h * HEAD_DIM;
  bf16x16 qf[4];
  #pragma unroll
  for (int c = 0; c < 4; c++) qf[c] = load_fragA(qbase + c * 32, QKV_LD, lane);

  const f32x8 z8 = {0.f,0.f,0.f,0.f,0.f,0.f,0.f,0.f};
  f32x8 o[8];
  float mrow[8], lrow[8];
  #pragma unroll
  for (int r = 0; r < 8; r++) { o[r] = z8; mrow[r] = -3.0e38f; lrow[r] = 0.f; }

  const float SCL = 0.08838834764831845f * 1.44269504088896f;   // 1/sqrt(128) * log2(e)
  const __bf16* kcol = qkv + (size_t)b * SEQ * QKV_LD + D_MODEL + kvh * HEAD_DIM;
  const __bf16* vb0  = vT  + (size_t)((b * N_KV_HEADS + kvh) * HEAD_DIM) * SEQ;

  auto tile = [&](int s0) {
    f32x8 sc0 = z8, sc1 = z8;
    const __bf16* kb = kcol + (size_t)s0 * QKV_LD;
    #pragma unroll
    for (int c = 0; c < 4; c++) {
      bf16x16 kf0 = load_fragB(kb + c * 32, QKV_LD, lane);
      bf16x16 kf1 = load_fragB(kb + (size_t)16 * QKV_LD + c * 32, QKV_LD, lane);
      sc0 = wmma_bf16(qf[c], kf0, sc0);
      sc1 = wmma_bf16(qf[c], kf1, sc1);
    }
    #pragma unroll
    for (int r = 0; r < 8; r++) {
      const int m  = m0 + hlf * 8 + r;
      const int n0 = s0 + colb;
      const int n1 = n0 + 16;
      const bool ok0 = (n0 <= m) && ((n0 < SINK) || (n0 >= m - (WINDOW - 1)));
      const bool ok1 = (n1 <= m) && ((n1 < SINK) || (n1 >= m - (WINDOW - 1)));
      float v0 = ok0 ? sc0[r] * SCL : -3.0e38f;
      float v1 = ok1 ? sc1[r] * SCL : -3.0e38f;
      float tmax = fmaxf(v0, v1);
      #pragma unroll
      for (int off = 1; off < 16; off <<= 1) tmax = fmaxf(tmax, __shfl_xor(tmax, off, 16));
      const float mnew  = fmaxf(mrow[r], tmax);
      const float alpha = exp2f(mrow[r] - mnew);
      mrow[r] = mnew;
      const float p0 = exp2f(v0 - mnew);
      const float p1 = exp2f(v1 - mnew);
      float ssum = p0 + p1;
      #pragma unroll
      for (int off = 1; off < 16; off <<= 1) ssum += __shfl_xor(ssum, off, 16);
      lrow[r] = lrow[r] * alpha + ssum;
      #pragma unroll
      for (int f = 0; f < 8; f++) o[f][r] *= alpha;
      pl[(hlf * 8 + r) * 40 + colb]      = (__bf16)p0;
      pl[(hlf * 8 + r) * 40 + 16 + colb] = (__bf16)p1;
    }
    bf16x16 pf = load_fragA(pl, 40, lane);
    const __bf16* vb = vb0 + s0;
    #pragma unroll
    for (int nd = 0; nd < 8; nd++) {
      bf16x16 vf = load_fragB(vb + (size_t)(nd * 16) * SEQ, SEQ, lane);
      o[nd] = wmma_bf16(pf, vf, o[nd]);
    }
  };

  int wstart = m0 - (WINDOW - 1); if (wstart < 0) wstart = 0;
  const int ts = wstart & ~31;
  if (ts > 0) tile(0);                              // sink-only tile
  for (int s0 = ts; s0 <= m0 + 15; s0 += 32) tile(s0);

  __bf16* ob = attnO + (size_t)(b * SEQ + m0) * (N_HEADS * HEAD_DIM) + h * HEAD_DIM;
  #pragma unroll
  for (int r = 0; r < 8; r++) {
    const float inv  = 1.0f / fmaxf(lrow[r], 1e-9f);
    const int   mloc = hlf * 8 + r;
    #pragma unroll
    for (int nd = 0; nd < 8; nd++)
      ob[(size_t)mloc * (N_HEADS * HEAD_DIM) + nd * 16 + colb] = (__bf16)(o[nd][r] * inv);
  }
}

extern "C" void kernel_launch(void* const* d_in, const int* in_sizes, int n_in,
                              void* d_out, int out_size, void* d_ws, size_t ws_size,
                              hipStream_t stream) {
  const float* x  = (const float*)d_in[0];
  const float* wq = (const float*)d_in[1];
  const float* wk = (const float*)d_in[2];
  const float* wv = (const float*)d_in[3];
  const float* wo = (const float*)d_in[4];
  float* out = (float*)d_out;

  // Workspace layout (bf16 elements), total exactly 64 MB:
  __bf16* xb    = (__bf16*)d_ws;                                  // 4096 x 2048
  __bf16* wqkvb = xb    + (size_t)MTOT * D_MODEL;                 // 3072 x 2048 (wq|wk|wv)
  __bf16* wob   = wqkvb + (size_t)QKV_LD * D_MODEL;               // 2048 x 2048
  __bf16* qkv   = wob   + (size_t)D_MODEL * D_MODEL;              // 4096 x 3072
  __bf16* vTb   = qkv   + (size_t)MTOT * QKV_LD;                  // 2*4*128*2048
  __bf16* attn  = xb;   // alias: x consumed by GEMM1 before attn is written

  // f32 -> bf16 conversions
  cvt_bf16<<<(MTOT * D_MODEL / 8 + 255) / 256, 256, 0, stream>>>(x, xb, MTOT * D_MODEL / 8);
  cvt_bf16<<<(D_MODEL * D_MODEL / 8 + 255) / 256, 256, 0, stream>>>(wq, wqkvb, D_MODEL * D_MODEL / 8);
  cvt_bf16<<<(512 * D_MODEL / 8 + 255) / 256, 256, 0, stream>>>(wk, wqkvb + (size_t)D_MODEL * D_MODEL, 512 * D_MODEL / 8);
  cvt_bf16<<<(512 * D_MODEL / 8 + 255) / 256, 256, 0, stream>>>(wv, wqkvb + (size_t)(D_MODEL + 512) * D_MODEL, 512 * D_MODEL / 8);
  cvt_bf16<<<(D_MODEL * D_MODEL / 8 + 255) / 256, 256, 0, stream>>>(wo, wob, D_MODEL * D_MODEL / 8);

  // Fused QKV projection: [4096,2048] @ [3072,2048]^T -> bf16 [4096,3072]
  gemm_tdm<__bf16><<<dim3(QKV_LD / 128, MTOT / 128), 256, 0, stream>>>(
      xb, wqkvb, MTOT, QKV_LD, D_MODEL, qkv);

  transpose_v<<<(BATCH * N_KV_HEADS * HEAD_DIM * SEQ) / 256, 256, 0, stream>>>(qkv, vTb);

  attn_kernel<<<(BATCH * N_HEADS * (SEQ / 16)) / 8, 256, 0, stream>>>(qkv, vTb, attn);

  // Output projection: [4096,2048] @ [2048,2048]^T -> f32 out
  gemm_tdm<float><<<dim3(D_MODEL / 128, MTOT / 128), 256, 0, stream>>>(
      attn, wob, MTOT, D_MODEL, D_MODEL, out);
}
